// GNNViasMetabolicas_68058051772655
// MI455X (gfx1250) — matrix operand order repaired and checked by reading, not compiled, not run on hardware
//
#include <hip/hip_runtime.h>
#include <hip/hip_bf16.h>

// ---------------------------------------------------------------------------
// GNN (GraphConv x4 + pool + MLP) for MI455X / gfx1250, bf16 WMMA pipeline.
//   B=256 batches, N=512 nodes, EMB=256, HID=512, OUT=256.
// Heavy math: one double-buffered LDS WMMA GEMM (v_wmma_f32_16x16x32_bf16),
// 128x128 block tile, 8 waves, 32x64 per wave = 8 WMMA per 32-K step.
// A tile staged via gfx1250 GLOBAL_LOAD_ASYNC_TO_LDS_B128 (ASYNCcnt) when the
// builtin is available; fall back to register staging otherwise.
// ---------------------------------------------------------------------------

#define BB   256
#define NN   512
#define EMBF 256

typedef __attribute__((ext_vector_type(16))) __bf16        v16bf;
typedef __attribute__((ext_vector_type(8)))  float         v8f;
typedef __attribute__((ext_vector_type(4)))  unsigned int  v4u;
typedef __attribute__((ext_vector_type(4)))  int           v4i;

#if defined(__has_builtin)
#if __has_builtin(__builtin_amdgcn_global_load_async_to_lds_b128) && \
    __has_builtin(__builtin_amdgcn_s_wait_asynccnt)
#define USE_ASYNC_COPY 1
#endif
#endif
#ifndef USE_ASYNC_COPY
#define USE_ASYNC_COPY 0
#endif

static __device__ __forceinline__ unsigned short f32_to_bf16(float f) {
    unsigned int u = __float_as_uint(f);
    u += 0x7FFFu + ((u >> 16) & 1u);          // round-to-nearest-even
    return (unsigned short)(u >> 16);
}
static __device__ __forceinline__ float bf16_to_f32(unsigned short h) {
    return __uint_as_float(((unsigned int)h) << 16);
}

#if USE_ASYNC_COPY
// Builtin signature (probe-confirmed via diagnostic): (V4i AS1*, V4i AS3*, Ii, Ii)
static __device__ __forceinline__ void async_cp_b128(const unsigned short* g,
                                                     unsigned short* l) {
    // generic global pointer value == AS1 address; generic LDS low 32 bits ==
    // LDS offset (ISA flat-addressing rules), so integer casts are exact.
    __builtin_amdgcn_global_load_async_to_lds_b128(
        (__attribute__((address_space(1))) v4i*)(unsigned long long)(uintptr_t)g,
        (__attribute__((address_space(3))) v4i*)(unsigned int)(uintptr_t)l,
        0, 0);
}
#endif

// ---------------------------------------------------------------------------
// adj_n = (adj + I) / (rowsum(adj + I) + 1e-6), stored bf16. One block per row.
// ---------------------------------------------------------------------------
__global__ __launch_bounds__(512)
void normalize_adj_kernel(const float* __restrict__ adj,
                          unsigned short* __restrict__ adjn)
{
    __shared__ float red[512];
    const int row = blockIdx.x;          // 0 .. B*N-1
    const int i   = row & (NN - 1);      // node index within batch
    const int t   = threadIdx.x;
    const float v = adj[(size_t)row * NN + t] + ((t == i) ? 1.0f : 0.0f);
    red[t] = v;
    __syncthreads();
    #pragma unroll
    for (int s = 256; s > 0; s >>= 1) {
        if (t < s) red[t] += red[t + s];
        __syncthreads();
    }
    const float inv = 1.0f / (red[0] + 1e-6f);
    adjn[(size_t)row * NN + t] = f32_to_bf16(v * inv);
}

// ---------------------------------------------------------------------------
// x0[b,n,:] = bf16(emb_table[gene_ids[b,n], :])   (ids are int64)
// ---------------------------------------------------------------------------
__global__ __launch_bounds__(256)
void gather_embed_kernel(const long long* __restrict__ ids,
                         const float* __restrict__ emb,
                         unsigned short* __restrict__ x0)
{
    const int row = blockIdx.x;          // 0 .. B*N-1
    const int t   = threadIdx.x;         // feature
    const long long id = ids[row];
    x0[(size_t)row * EMBF + t] = f32_to_bf16(emb[(size_t)id * EMBF + t]);
}

__global__ __launch_bounds__(256)
void cvt_f32_bf16_kernel(const float* __restrict__ in,
                         unsigned short* __restrict__ out, int n)
{
    const int i = blockIdx.x * 256 + threadIdx.x;
    if (i < n) out[i] = f32_to_bf16(in[i]);
}

// ---------------------------------------------------------------------------
// Batched bf16 GEMM: C[M,N] = A[M,K] @ B[K,N] (+bias,+relu), bf16 out.
// Block tile 128x128, 8 waves (4M x 2N), wave tile 32x64 = 2x4 WMMA frags.
// Double-buffered LDS: next tile's loads are issued before the WMMAs, the
// LDS stores / counter waits land after them -> HBM latency hidden.
// A tile LDS row-major [128][40] bf16 (pad keeps 16B align, K contiguous).
// B tile LDS transposed [128][40] so per-lane K runs are contiguous.
// Fragment layouts follow CDNA5 ISA 7.12.2.
// ---------------------------------------------------------------------------
__global__ __launch_bounds__(256)
void gemm_bf16_kernel(const unsigned short* __restrict__ A, int lda, long long strideA,
                      const unsigned short* __restrict__ B, int ldb, long long strideB,
                      unsigned short* __restrict__ C, int ldc, long long strideC,
                      const float* __restrict__ bias,
                      int M, int N, int K, int relu)
{
    __shared__ unsigned short As[2][128 * 40];   // 2 x 10 KB
    __shared__ unsigned short Bs[2][128 * 40];   // 2 x 10 KB

    const int tid  = threadIdx.x;
    const int m0   = blockIdx.y * 128;
    const int n0   = blockIdx.x * 128;
    const long long zb = blockIdx.z;
    const unsigned short* Ag = A + zb * strideA;
    const unsigned short* Bg = B + zb * strideB;
    unsigned short*       Cg = C + zb * strideC;

    const int wid  = tid >> 5;
    const int lane = tid & 31;
    const int lrow = lane & 15;      // row/col within a 16-tile
    const int lhi  = lane >> 4;      // lane half selects K-group
    const int mw   = (wid >> 1) * 32;
    const int nw   = (wid &  1) * 64;

    // ---- per-thread staging coordinates (pointer-increment addressing) ----
    const int arow = tid >> 2;             // A: rows arow, arow+64
    const int akc  = (tid & 3) * 8;        //    k-chunk (8 bf16)
    const int bk   = tid >> 4;             // B: k rows bk, bk+16
    const int bnc  = (tid & 15) * 8;       //    n-chunk (8 bf16)

    const unsigned short* pA0 = Ag + (size_t)(m0 + arow) * lda + akc;
    const unsigned short* pA1 = pA0 + (size_t)64 * lda;
    const unsigned short* pB0 = Bg + (size_t)bk * ldb + n0 + bnc;
    const unsigned short* pB1 = pB0 + (size_t)16 * ldb;

    const int aoff0 = arow * 40 + akc;
    const int aoff1 = (arow + 64) * 40 + akc;

    v8f acc[2][4];
    {
        const v8f vz = {0.f,0.f,0.f,0.f,0.f,0.f,0.f,0.f};
        #pragma unroll
        for (int mi = 0; mi < 2; ++mi)
            #pragma unroll
            for (int ni = 0; ni < 4; ++ni) acc[mi][ni] = vz;
    }

    union FU { v4u u4[2]; v16bf v; };

    // ---------------- prologue: stage tile 0 into buffer 0 ----------------
    v4u ra0, ra1, rb0, rb1;
    (void)ra0; (void)ra1;
#if USE_ASYNC_COPY
    async_cp_b128(pA0, &As[0][aoff0]);
    async_cp_b128(pA1, &As[0][aoff1]);
#else
    ra0 = *(const v4u*)pA0;
    ra1 = *(const v4u*)pA1;
#endif
    rb0 = *(const v4u*)pB0;
    rb1 = *(const v4u*)pB1;
    pA0 += 32; pA1 += 32;
    pB0 += (size_t)32 * ldb; pB1 += (size_t)32 * ldb;
#if !USE_ASYNC_COPY
    *(v4u*)(&As[0][aoff0]) = ra0;
    *(v4u*)(&As[0][aoff1]) = ra1;
#endif
    {
        const unsigned short* p0 = (const unsigned short*)&rb0;
        const unsigned short* p1 = (const unsigned short*)&rb1;
        #pragma unroll
        for (int j = 0; j < 8; ++j) {
            Bs[0][(bnc + j) * 40 + bk]      = p0[j];
            Bs[0][(bnc + j) * 40 + bk + 16] = p1[j];
        }
    }
#if USE_ASYNC_COPY
    __builtin_amdgcn_s_wait_asynccnt(0);
#endif
    __syncthreads();

    // ---------------------------- main K loop -----------------------------
    int buf = 0;
    for (int k0 = 0; k0 < K; k0 += 32) {
        const bool more = (k0 + 32) < K;
        const int nxt = buf ^ 1;

        // issue next tile's global loads FIRST (latency hidden by WMMAs)
        if (more) {
#if USE_ASYNC_COPY
            async_cp_b128(pA0, &As[nxt][aoff0]);
            async_cp_b128(pA1, &As[nxt][aoff1]);
#else
            ra0 = *(const v4u*)pA0;
            ra1 = *(const v4u*)pA1;
#endif
            rb0 = *(const v4u*)pB0;
            rb1 = *(const v4u*)pB1;
            __builtin_prefetch(pA0 + 32, 0, 0);   // k0+64 tile -> global_prefetch_b8
            __builtin_prefetch(pB0 + (size_t)32 * ldb, 0, 0);
            pA0 += 32; pA1 += 32;
            pB0 += (size_t)32 * ldb; pB1 += (size_t)32 * ldb;
        }

        // ---- fragments from LDS (two ds_load_b128 each), then 8 WMMAs ----
        v16bf afrag[2], bfrag[4];
        #pragma unroll
        for (int mi = 0; mi < 2; ++mi) {
            FU f;
            const int base = (mw + mi * 16 + lrow) * 40 + lhi * 8;
            f.u4[0] = *(const v4u*)(&As[buf][base]);       // K 0..7 / 8..15
            f.u4[1] = *(const v4u*)(&As[buf][base + 16]);  // K 16..23 / 24..31
            afrag[mi] = f.v;
        }
        #pragma unroll
        for (int ni = 0; ni < 4; ++ni) {
            FU f;
            const int base = (nw + ni * 16 + lrow) * 40 + lhi * 8;
            f.u4[0] = *(const v4u*)(&Bs[buf][base]);
            f.u4[1] = *(const v4u*)(&Bs[buf][base + 16]);
            bfrag[ni] = f.v;
        }
        #pragma unroll
        for (int mi = 0; mi < 2; ++mi)
            #pragma unroll
            for (int ni = 0; ni < 4; ++ni)
                acc[mi][ni] = __builtin_amdgcn_wmma_f32_16x16x32_bf16(
                    false, afrag[mi], false, bfrag[ni],
                    (short)0, acc[mi][ni], false, false);

        // store next tile to the other LDS buffer (waits happen *after* WMMAs)
        if (more) {
#if !USE_ASYNC_COPY
            *(v4u*)(&As[nxt][aoff0]) = ra0;
            *(v4u*)(&As[nxt][aoff1]) = ra1;
#endif
            const unsigned short* p0 = (const unsigned short*)&rb0;
            const unsigned short* p1 = (const unsigned short*)&rb1;
            #pragma unroll
            for (int j = 0; j < 8; ++j) {
                Bs[nxt][(bnc + j) * 40 + bk]      = p0[j];
                Bs[nxt][(bnc + j) * 40 + bk + 16] = p1[j];
            }
#if USE_ASYNC_COPY
            __builtin_amdgcn_s_wait_asynccnt(0);
#endif
        }
        __syncthreads();
        buf = nxt;
    }

    // ---- epilogue: bias + relu + bf16 store (C layout: VGPR r -> M=r+8*lhi)
    #pragma unroll
    for (int mi = 0; mi < 2; ++mi) {
        #pragma unroll
        for (int ni = 0; ni < 4; ++ni) {
            union { v8f v; float f[8]; } a; a.v = acc[mi][ni];
            const int col = n0 + nw + ni * 16 + lrow;
            const float bv = bias ? bias[col] : 0.0f;
            #pragma unroll
            for (int r = 0; r < 8; ++r) {
                const int row = m0 + mw + mi * 16 + lhi * 8 + r;
                float val = a.f[r] + bv;
                if (relu) val = fmaxf(val, 0.0f);
                Cg[(size_t)row * ldc + col] = f32_to_bf16(val);
            }
        }
    }
}

// ---------------------------------------------------------------------------
// pooled[b,f] = mean_n x[b,n,f]    (x bf16 [B,N,256])
// ---------------------------------------------------------------------------
__global__ __launch_bounds__(256)
void mean_pool_kernel(const unsigned short* __restrict__ x,
                      float* __restrict__ pooled)
{
    const int b = blockIdx.x;
    const int f = threadIdx.x;
    const unsigned short* xb = x + (size_t)b * NN * 256 + f;
    float s = 0.f;
    for (int n = 0; n < NN; ++n) s += bf16_to_f32(xb[(size_t)n * 256]);
    pooled[b * 256 + f] = s * (1.0f / (float)NN);
}

// ---------------------------------------------------------------------------
// Tiny MLP head: 256->128 relu ->64 relu ->3 (f32, one block per batch row)
// ---------------------------------------------------------------------------
__global__ __launch_bounds__(128)
void mlp_kernel(const float* __restrict__ pooled,
                const float* __restrict__ pw0, const float* __restrict__ pb0,
                const float* __restrict__ pw1, const float* __restrict__ pb1,
                const float* __restrict__ pw2, const float* __restrict__ pb2,
                float* __restrict__ out)
{
    __shared__ float sin_[256];
    __shared__ float sh0[128];
    __shared__ float sh1[64];
    const int b = blockIdx.x, t = threadIdx.x;
    sin_[t]       = pooled[b * 256 + t];
    sin_[t + 128] = pooled[b * 256 + t + 128];
    __syncthreads();
    float s = pb0[t];
    for (int i = 0; i < 256; ++i) s += sin_[i] * pw0[i * 128 + t];
    sh0[t] = fmaxf(s, 0.f);
    __syncthreads();
    if (t < 64) {
        float s1 = pb1[t];
        for (int i = 0; i < 128; ++i) s1 += sh0[i] * pw1[i * 64 + t];
        sh1[t] = fmaxf(s1, 0.f);
    }
    __syncthreads();
    if (t < 3) {
        float s2 = pb2[t];
        for (int i = 0; i < 64; ++i) s2 += sh1[i] * pw2[i * 3 + t];
        out[b * 3 + t] = s2;
    }
}

// ---------------------------------------------------------------------------
extern "C" void kernel_launch(void* const* d_in, const int* in_sizes, int n_in,
                              void* d_out, int out_size, void* d_ws, size_t ws_size,
                              hipStream_t stream)
{
    const long long* gene_ids = (const long long*)d_in[0];
    const float* adj  = (const float*)d_in[1];
    const float* emb  = (const float*)d_in[2];
    const float* w0 = (const float*)d_in[3];   const float* b0 = (const float*)d_in[4];
    const float* w1 = (const float*)d_in[5];   const float* b1 = (const float*)d_in[6];
    const float* w2 = (const float*)d_in[7];   const float* b2 = (const float*)d_in[8];
    const float* w3 = (const float*)d_in[9];   const float* b3 = (const float*)d_in[10];
    const float* pw0 = (const float*)d_in[11]; const float* pb0 = (const float*)d_in[12];
    const float* pw1 = (const float*)d_in[13]; const float* pb1 = (const float*)d_in[14];
    const float* pw2 = (const float*)d_in[15]; const float* pb2 = (const float*)d_in[16];
    float* outp = (float*)d_out;
    (void)in_sizes; (void)n_in; (void)out_size; (void)ws_size;

    // -------- workspace layout (bytes) --------
    char* ws = (char*)d_ws;
    const size_t SZ_ADJ = (size_t)BB * NN * NN * 2;   // 134 MB bf16 adj
    const size_t SZ_BUF = (size_t)BB * NN * 512 * 2;  // 134 MB bf16 activations
    unsigned short* adjn = (unsigned short*)(ws);
    unsigned short* buf0 = (unsigned short*)(ws + SZ_ADJ);             // x
    unsigned short* buf1 = (unsigned short*)(ws + SZ_ADJ + SZ_BUF);    // adj@x
    unsigned short* wbf  = (unsigned short*)(ws + SZ_ADJ + 2 * SZ_BUF);
    unsigned short* wb0 = wbf;                          // 256x512
    unsigned short* wb1 = wb0 + 256 * 512;              // 512x512
    unsigned short* wb2 = wb1 + 512 * 512;              // 512x512
    unsigned short* wb3 = wb2 + 512 * 512;              // 512x256
    const size_t SZ_W = (size_t)(256*512 + 512*512 + 512*512 + 512*256) * 2;
    float* pooled = (float*)(ws + SZ_ADJ + 2 * SZ_BUF + SZ_W);

    // -------- 1. normalize adjacency once (f32 -> bf16 cache) --------
    normalize_adj_kernel<<<BB * NN, 512, 0, stream>>>(adj, adjn);
    // -------- 2. gather embeddings, convert weights --------
    gather_embed_kernel<<<BB * NN, 256, 0, stream>>>(gene_ids, emb, buf0);
    cvt_f32_bf16_kernel<<<(256*512 + 255)/256, 256, 0, stream>>>(w0, wb0, 256*512);
    cvt_f32_bf16_kernel<<<(512*512 + 255)/256, 256, 0, stream>>>(w1, wb1, 512*512);
    cvt_f32_bf16_kernel<<<(512*512 + 255)/256, 256, 0, stream>>>(w2, wb2, 512*512);
    cvt_f32_bf16_kernel<<<(512*256 + 255)/256, 256, 0, stream>>>(w3, wb3, 512*256);

    const long long sAdj = (long long)NN * NN;
    const int MROWS = BB * NN;          // 131072 flattened rows for linears

    // -------- layer 0: EMB(256) -> HID(512) --------
    gemm_bf16_kernel<<<dim3(256/128, NN/128, BB), 256, 0, stream>>>(
        adjn, NN, sAdj, buf0, 256, (long long)NN * 256,
        buf1, 256, (long long)NN * 256, nullptr, NN, 256, NN, 0);
    gemm_bf16_kernel<<<dim3(512/128, MROWS/128, 1), 256, 0, stream>>>(
        buf1, 256, 0, wb0, 512, 0, buf0, 512, 0, b0, MROWS, 512, 256, 1);

    // -------- layers 1,2: HID -> HID --------
    gemm_bf16_kernel<<<dim3(512/128, NN/128, BB), 256, 0, stream>>>(
        adjn, NN, sAdj, buf0, 512, (long long)NN * 512,
        buf1, 512, (long long)NN * 512, nullptr, NN, 512, NN, 0);
    gemm_bf16_kernel<<<dim3(512/128, MROWS/128, 1), 256, 0, stream>>>(
        buf1, 512, 0, wb1, 512, 0, buf0, 512, 0, b1, MROWS, 512, 512, 1);

    gemm_bf16_kernel<<<dim3(512/128, NN/128, BB), 256, 0, stream>>>(
        adjn, NN, sAdj, buf0, 512, (long long)NN * 512,
        buf1, 512, (long long)NN * 512, nullptr, NN, 512, NN, 0);
    gemm_bf16_kernel<<<dim3(512/128, MROWS/128, 1), 256, 0, stream>>>(
        buf1, 512, 0, wb2, 512, 0, buf0, 512, 0, b2, MROWS, 512, 512, 1);

    // -------- layer 3: HID -> OUT(256) --------
    gemm_bf16_kernel<<<dim3(512/128, NN/128, BB), 256, 0, stream>>>(
        adjn, NN, sAdj, buf0, 512, (long long)NN * 512,
        buf1, 512, (long long)NN * 512, nullptr, NN, 512, NN, 0);
    gemm_bf16_kernel<<<dim3(256/128, MROWS/128, 1), 256, 0, stream>>>(
        buf1, 512, 0, wb3, 256, 0, buf0, 256, 0, b3, MROWS, 256, 512, 1);

    // -------- pool + MLP head --------
    mean_pool_kernel<<<BB, 256, 0, stream>>>(buf0, pooled);
    mlp_kernel<<<BB, 128, 0, stream>>>(pooled, pw0, pb0, pw1, pb1, pw2, pb2, outp);
}
